// FlashInferMoELayer_81973745811686
// MI455X (gfx1250) — matrix operand order repaired and checked by reading, not compiled, not run on hardware
//
#include <hip/hip_runtime.h>
#include <hip/hip_bf16.h>

typedef __bf16 bf16_t;
typedef __attribute__((ext_vector_type(16))) __bf16 v16bf;
typedef __attribute__((ext_vector_type(8)))  float  v8f;

#define T_TOK   4096
#define HDIM    1024
#define EXP     8
#define IDIM    2048
#define TWOI    4096

#define TILE_M      64
#define ACT_STRIDE  (IDIM + 16)   // bf16 elements; 16B-aligned rows, staggered banks

union FragU { v16bf v; uint4 q[2]; };

__device__ __forceinline__ float silu_fast(float g) {
    // g * sigmoid(g); v_rcp_f32 instead of IEEE divide (result stored as bf16)
    return g * __builtin_amdgcn_rcpf(1.0f + __expf(-g));
}

// ---------------------------------------------------------------------------
// Zero output + expert counters (harness poisons buffers; we accumulate).
// ---------------------------------------------------------------------------
__global__ void zero_init_kernel(float* __restrict__ out, int n, int* __restrict__ counts) {
    int i = blockIdx.x * blockDim.x + threadIdx.x;
    int stride = gridDim.x * blockDim.x;
    for (; i < n; i += stride) out[i] = 0.0f;
    if (blockIdx.x == 0 && threadIdx.x < EXP) counts[threadIdx.x] = 0;
}

// ---------------------------------------------------------------------------
// Router: logits -> softmax -> top2 -> normalized weights + compacted
// per-expert token lists. Also converts the token's x row to bf16.
// One block (128 threads) per token.
// ---------------------------------------------------------------------------
__global__ void router_kernel(const float* __restrict__ x,
                              const float* __restrict__ rw,
                              bf16_t* __restrict__ xb,
                              float* __restrict__ wmat,
                              int* __restrict__ idx,
                              int* __restrict__ counts) {
    __shared__ float red[EXP][128];
    const int token = blockIdx.x;
    const int tid   = threadIdx.x;
    const float* xrow = x + (size_t)token * HDIM;

    float acc[EXP];
#pragma unroll
    for (int e = 0; e < EXP; ++e) acc[e] = 0.0f;

    for (int i = tid; i < HDIM; i += 128) {
        float xv = xrow[i];
        xb[(size_t)token * HDIM + i] = (bf16_t)xv;
#pragma unroll
        for (int e = 0; e < EXP; ++e) acc[e] += xv * rw[e * HDIM + i];
    }
#pragma unroll
    for (int e = 0; e < EXP; ++e) red[e][tid] = acc[e];
    __syncthreads();
    for (int s = 64; s > 0; s >>= 1) {
        if (tid < s) {
#pragma unroll
            for (int e = 0; e < EXP; ++e) red[e][tid] += red[e][tid + s];
        }
        __syncthreads();
    }
    if (tid == 0) {
        float lg[EXP];
        float mx = -1e30f;
#pragma unroll
        for (int e = 0; e < EXP; ++e) { lg[e] = red[e][0]; mx = fmaxf(mx, lg[e]); }
        float sum = 0.0f;
#pragma unroll
        for (int e = 0; e < EXP; ++e) { lg[e] = __expf(lg[e] - mx); sum += lg[e]; }
        float inv = 1.0f / sum;
        int b1 = 0; float p1 = -1.0f;
#pragma unroll
        for (int e = 0; e < EXP; ++e) { float p = lg[e] * inv; lg[e] = p; if (p > p1) { p1 = p; b1 = e; } }
        int b2 = 0; float p2 = -1.0f;
#pragma unroll
        for (int e = 0; e < EXP; ++e) if (e != b1 && lg[e] > p2) { p2 = lg[e]; b2 = e; }
        float norm = 1.0f / (p1 + p2);
#pragma unroll
        for (int e = 0; e < EXP; ++e) wmat[token * EXP + e] = 0.0f;
        wmat[token * EXP + b1] = p1 * norm;
        wmat[token * EXP + b2] = p2 * norm;
        int s1 = atomicAdd(&counts[b1], 1); idx[b1 * T_TOK + s1] = token;
        int s2 = atomicAdd(&counts[b2], 1); idx[b2 * T_TOK + s2] = token;
    }
}

// ---------------------------------------------------------------------------
// Convert expert weights fp32 -> bf16 (bf16 weights ~96 MB: L2 resident).
// ---------------------------------------------------------------------------
__global__ void convert_w_kernel(const float* __restrict__ w1, const float* __restrict__ w2,
                                 bf16_t* __restrict__ w1b, bf16_t* __restrict__ w2b) {
    size_t i = blockIdx.x * (size_t)blockDim.x + threadIdx.x;
    size_t stride = gridDim.x * (size_t)blockDim.x;
    const size_t n1 = (size_t)EXP * TWOI * HDIM;
    const size_t n2 = (size_t)EXP * HDIM * IDIM;
    for (size_t j = i; j < n1; j += stride) w1b[j] = (bf16_t)w1[j];
    for (size_t j = i; j < n2; j += stride) w2b[j] = (bf16_t)w2[j];
}

// ---------------------------------------------------------------------------
// Fused expert FFN over a 64-token tile. 256 threads = 8 wave32.
// LDS (264KB act tile) limits us to 1 block/WGP anyway, so __launch_bounds__
// (256, 1) releases the full VGPR budget — 16 f32 accumulators + 4 B frags
// stay resident with no spills.
//   Phase 1: each wave computes gate AND up for 32 columns (4 B frags,
//            16 accumulators); A fragments amortized over 16 WMMAs;
//            silu(g)*u in registers -> bf16 act in LDS; no inner barriers.
//   Phase 2: y = act @ w2^T from LDS act, four N-tiles per wave pass;
//            weighted atomicAdd scatter to out.
// All WMMA sites are block-uniform (EXEC all ones).
// ---------------------------------------------------------------------------
__global__ void __launch_bounds__(256, 1)
moe_expert_kernel(const bf16_t* __restrict__ xb,
                  const bf16_t* __restrict__ w1b,
                  const bf16_t* __restrict__ w2b,
                  const float* __restrict__ wmat,
                  const int* __restrict__ idx,
                  const int* __restrict__ counts,
                  float* __restrict__ out) {
    extern __shared__ char smem_raw[];
    bf16_t* actLDS = (bf16_t*)smem_raw;              // 64 * ACT_STRIDE bf16 = 264KB
    __shared__ int   s_tok[TILE_M];
    __shared__ float s_wt[TILE_M];

    const int e    = blockIdx.y;
    const int tile = blockIdx.x;
    const int cnt  = counts[e];
    const int numTiles = (cnt + TILE_M - 1) / TILE_M;
    if (tile >= numTiles) return;   // uniform per block, before any barrier

    const int tid  = threadIdx.x;
    const int lane = tid & 31;
    const int wave = tid >> 5;
    const int mrow = lane & 15;     // M row (A) or N col (B/C) within a 16-tile
    const int kb   = lane >> 4;     // K-half select (this code's contiguous k = 16*kb + s order)

    if (tid < TILE_M) {
        int slot = tile * TILE_M + tid;
        if (slot < cnt) {
            int t = idx[e * T_TOK + slot];
            s_tok[tid] = t;
            s_wt[tid]  = wmat[t * EXP + e];
        } else { s_tok[tid] = 0; s_wt[tid] = 0.0f; }
    }
    __syncthreads();

    const bf16_t* w1e = w1b + (size_t)e * TWOI * HDIM;
    const bf16_t* w2e = w2b + (size_t)e * HDIM * IDIM;

    // Hoisted A-row base pointers (loop-invariant across K and chunks).
    const bf16_t* aBase[4];
#pragma unroll
    for (int sub = 0; sub < 4; ++sub)
        aBase[sub] = xb + (size_t)s_tok[sub * 16 + mrow] * HDIM + kb * 16;

    // ---------------- Phase 1: act tile in LDS (no inner barriers) ----------------
    for (int c = 0; c < IDIM; c += 256) {
        const int col0 = c + wave * 32;                 // this wave's 32 columns
        const bf16_t* bg0Row = w1e + (size_t)(col0 + mrow) * HDIM + kb * 16;
        const bf16_t* bg1Row = bg0Row + (size_t)16 * HDIM;
        const bf16_t* bu0Row = bg0Row + (size_t)IDIM * HDIM;
        const bf16_t* bu1Row = bu0Row + (size_t)16 * HDIM;
        v8f accG0[4] = {}, accG1[4] = {};
        v8f accU0[4] = {}, accU1[4] = {};
        for (int k0 = 0; k0 < HDIM; k0 += 32) {
            FragU bG0, bG1, bU0, bU1;        // B 32x16: 16 contiguous K at k0 + kb*16
            __builtin_prefetch(bg0Row + k0 + 256, 0, 3);
            bG0.q[0] = *(const uint4*)(bg0Row + k0);
            bG0.q[1] = *(const uint4*)(bg0Row + k0 + 8);
            bG1.q[0] = *(const uint4*)(bg1Row + k0);
            bG1.q[1] = *(const uint4*)(bg1Row + k0 + 8);
            bU0.q[0] = *(const uint4*)(bu0Row + k0);
            bU0.q[1] = *(const uint4*)(bu0Row + k0 + 8);
            bU1.q[0] = *(const uint4*)(bu1Row + k0);
            bU1.q[1] = *(const uint4*)(bu1Row + k0 + 8);
#pragma unroll
            for (int sub = 0; sub < 4; ++sub) {
                FragU a;                     // A 16x32: 16 contiguous K at k0 + kb*16
                a.q[0] = *(const uint4*)(aBase[sub] + k0);
                a.q[1] = *(const uint4*)(aBase[sub] + k0 + 8);
                accG0[sub] = __builtin_amdgcn_wmma_f32_16x16x32_bf16(
                    false, a.v, false, bG0.v, (short)0, accG0[sub], false, false);
                accG1[sub] = __builtin_amdgcn_wmma_f32_16x16x32_bf16(
                    false, a.v, false, bG1.v, (short)0, accG1[sub], false, false);
                accU0[sub] = __builtin_amdgcn_wmma_f32_16x16x32_bf16(
                    false, a.v, false, bU0.v, (short)0, accU0[sub], false, false);
                accU1[sub] = __builtin_amdgcn_wmma_f32_16x16x32_bf16(
                    false, a.v, false, bU1.v, (short)0, accU1[sub], false, false);
            }
        }
        // gate/up land on identical (M,N) slots: silu in registers, store bf16.
        // C/D layout: VGPR r -> M = r + 8*kb ; col = lane&15.
#pragma unroll
        for (int sub = 0; sub < 4; ++sub)
#pragma unroll
            for (int r = 0; r < 8; ++r) {
                const int row = sub * 16 + r + 8 * kb;
                bf16_t* arow = actLDS + row * ACT_STRIDE + col0 + mrow;
                arow[0]  = (bf16_t)(silu_fast(accG0[sub][r]) * accU0[sub][r]);
                arow[16] = (bf16_t)(silu_fast(accG1[sub][r]) * accU1[sub][r]);
            }
    }
    __syncthreads();   // single barrier: act tile complete

    // ---------------- Phase 2: y = act @ w2^T, four N-tiles per pass ----------------
    for (int np = wave; np < HDIM / 64; np += 8) {
        const int n0 = np * 64;
        const bf16_t* bRow0 = w2e + (size_t)(n0 + mrow) * IDIM + kb * 16;
        const bf16_t* bRow1 = bRow0 + (size_t)16 * IDIM;
        const bf16_t* bRow2 = bRow0 + (size_t)32 * IDIM;
        const bf16_t* bRow3 = bRow0 + (size_t)48 * IDIM;
        v8f acc0[4] = {}, acc1[4] = {}, acc2[4] = {}, acc3[4] = {};
        for (int k0 = 0; k0 < IDIM; k0 += 32) {
            FragU b0, b1, b2, b3;
            __builtin_prefetch(bRow0 + k0 + 256, 0, 3);
            b0.q[0] = *(const uint4*)(bRow0 + k0);
            b0.q[1] = *(const uint4*)(bRow0 + k0 + 8);
            b1.q[0] = *(const uint4*)(bRow1 + k0);
            b1.q[1] = *(const uint4*)(bRow1 + k0 + 8);
            b2.q[0] = *(const uint4*)(bRow2 + k0);
            b2.q[1] = *(const uint4*)(bRow2 + k0 + 8);
            b3.q[0] = *(const uint4*)(bRow3 + k0);
            b3.q[1] = *(const uint4*)(bRow3 + k0 + 8);
#pragma unroll
            for (int sub = 0; sub < 4; ++sub) {
                const bf16_t* ap = actLDS + (sub * 16 + mrow) * ACT_STRIDE + kb * 16 + k0;
                FragU a;
                a.q[0] = *(const uint4*)ap;
                a.q[1] = *(const uint4*)(ap + 8);
                acc0[sub] = __builtin_amdgcn_wmma_f32_16x16x32_bf16(
                    false, a.v, false, b0.v, (short)0, acc0[sub], false, false);
                acc1[sub] = __builtin_amdgcn_wmma_f32_16x16x32_bf16(
                    false, a.v, false, b1.v, (short)0, acc1[sub], false, false);
                acc2[sub] = __builtin_amdgcn_wmma_f32_16x16x32_bf16(
                    false, a.v, false, b2.v, (short)0, acc2[sub], false, false);
                acc3[sub] = __builtin_amdgcn_wmma_f32_16x16x32_bf16(
                    false, a.v, false, b3.v, (short)0, acc3[sub], false, false);
            }
        }
#pragma unroll
        for (int sub = 0; sub < 4; ++sub)
#pragma unroll
            for (int r = 0; r < 8; ++r) {
                const int row = sub * 16 + r + 8 * kb;
                const float wt = s_wt[row];
                if (wt != 0.0f) {
                    float* orow = out + (size_t)s_tok[row] * HDIM + n0 + mrow;
                    atomicAdd(orow,      wt * acc0[sub][r]);
                    atomicAdd(orow + 16, wt * acc1[sub][r]);
                    atomicAdd(orow + 32, wt * acc2[sub][r]);
                    atomicAdd(orow + 48, wt * acc3[sub][r]);
                }
            }
    }
}

// ---------------------------------------------------------------------------
extern "C" void kernel_launch(void* const* d_in, const int* in_sizes, int n_in,
                              void* d_out, int out_size, void* d_ws, size_t ws_size,
                              hipStream_t stream) {
    const float* x  = (const float*)d_in[0];
    const float* rw = (const float*)d_in[1];
    const float* w1 = (const float*)d_in[2];
    const float* w2 = (const float*)d_in[3];
    float* out = (float*)d_out;

    char* ws = (char*)d_ws;
    size_t off = 0;
    auto alloc = [&](size_t bytes) -> void* {
        void* p = ws + off;
        off = (off + bytes + 255) & ~(size_t)255;
        return p;
    };
    bf16_t* xb     = (bf16_t*)alloc((size_t)T_TOK * HDIM * 2);          //   8 MB
    bf16_t* w1b    = (bf16_t*)alloc((size_t)EXP * TWOI * HDIM * 2);     //  64 MB
    bf16_t* w2b    = (bf16_t*)alloc((size_t)EXP * HDIM * IDIM * 2);     //  32 MB
    float*  wmat   = (float*)alloc((size_t)T_TOK * EXP * 4);
    int*    idx    = (int*)alloc((size_t)EXP * T_TOK * 4);
    int*    counts = (int*)alloc(256);

    zero_init_kernel<<<2048, 256, 0, stream>>>(out, out_size, counts);
    router_kernel<<<T_TOK, 128, 0, stream>>>(x, rw, xb, wmat, idx, counts);
    convert_w_kernel<<<4096, 256, 0, stream>>>(w1, w2, w1b, w2b);

    const size_t shmem = (size_t)TILE_M * ACT_STRIDE * 2;   // act tile (bf16), 264KB
    dim3 grid(T_TOK / TILE_M, EXP);
    moe_expert_kernel<<<grid, 256, shmem, stream>>>(xb, w1b, w2b, wmat, idx, counts, out);
}
